// SparseUnet_43327630082529
// MI455X (gfx1250) — matrix operand order repaired and checked by reading, not compiled, not run on hardware
//
#include <hip/hip_runtime.h>
#include <cstdint>

typedef __attribute__((ext_vector_type(16))) _Float16 v16h;
typedef __attribute__((ext_vector_type(8)))  _Float16 v8h;
typedef __attribute__((ext_vector_type(8)))  float    v8f;

// ---------------------------------------------------------------------------
// Elementwise / utility kernels
// ---------------------------------------------------------------------------

__global__ __launch_bounds__(256) void k_fill_zero(uint32_t* __restrict__ p, long n) {
  long i = (long)blockIdx.x * blockDim.x + threadIdx.x;
  if (i < n) p[i] = 0u;
}

__global__ __launch_bounds__(256) void k_scatter(const float* __restrict__ vf,
                                                 const int* __restrict__ vc,
                                                 _Float16* __restrict__ x0,
                                                 float* __restrict__ m1, int N) {
  int i = blockIdx.x * blockDim.x + threadIdx.x;
  if (i >= N) return;
  int b = vc[i * 4 + 0], z = vc[i * 4 + 1], y = vc[i * 4 + 2], x = vc[i * 4 + 3];
  long vox = ((long)(b * 8 + z) * 48 + y) * 48 + x;
  m1[vox] = 1.0f;
#pragma unroll
  for (int c = 0; c < 4; ++c)
    x0[vox * 32 + c] = (_Float16)vf[i * 4 + c];
}

// Pack f32 weights (3,3,3,cin,cout) -> f16 [cout][27*cinP], optional spatial flip.
__global__ __launch_bounds__(256) void k_pack_w(const float* __restrict__ w,
                                                _Float16* __restrict__ wt,
                                                int cin, int cout, int cinP, int flip) {
  long idx = (long)blockIdx.x * blockDim.x + threadIdx.x;
  int Kt = 27 * cinP;
  long total = (long)cout * Kt;
  if (idx >= total) return;
  int n = (int)(idx / Kt);
  int r = (int)(idx - (long)n * Kt);
  int t = r / cinP;
  int c = r - t * cinP;
  float v = 0.0f;
  if (c < cin) {
    int ts = flip ? (26 - t) : t;
    v = w[((long)ts * cin + c) * cout + n];
  }
  wt[(long)n * Kt + r] = (_Float16)v;
}

__global__ __launch_bounds__(256) void k_maxpool_mask(const float* __restrict__ mi,
                                                      float* __restrict__ mo,
                                                      int Bn, int Di, int Hi, int Wi,
                                                      int Do, int Ho, int Wo) {
  int idx = blockIdx.x * blockDim.x + threadIdx.x;
  int tot = Bn * Do * Ho * Wo;
  if (idx >= tot) return;
  int t = idx;
  int x = t % Wo; t /= Wo;
  int y = t % Ho; t /= Ho;
  int z = t % Do; int b = t / Do;
  float v = 0.f;
  for (int dz = 0; dz < 3; ++dz) {
    int iz = 2 * z + dz - 1; if (iz < 0 || iz >= Di) continue;
    for (int dy = 0; dy < 3; ++dy) {
      int iy = 2 * y + dy - 1; if (iy < 0 || iy >= Hi) continue;
      for (int dx = 0; dx < 3; ++dx) {
        int ix = 2 * x + dx - 1; if (ix < 0 || ix >= Wi) continue;
        v = fmaxf(v, mi[((b * Di + iz) * Hi + iy) * Wi + ix]);
      }
    }
  }
  mo[idx] = v;
}

__global__ __launch_bounds__(256) void k_concat(const _Float16* __restrict__ A,
                                                const _Float16* __restrict__ Bs,
                                                _Float16* __restrict__ out,
                                                int C, long total) {
  long i = (long)blockIdx.x * blockDim.x + threadIdx.x;
  if (i >= total) return;
  int c2 = (int)(i % (2 * C));
  long v = i / (2 * C);
  out[i] = (c2 < C) ? A[v * C + c2] : Bs[v * C + (c2 - C)];
}

__global__ __launch_bounds__(64) void k_se_reduce(const _Float16* __restrict__ o,
                                                  float* __restrict__ s,
                                                  int vPerB, int C) {
  int c = blockIdx.x * 64 + threadIdx.x;
  int b = blockIdx.y;
  if (c >= C) return;
  const _Float16* p = o + (long)b * vPerB * C + c;
  float acc = 0.f;
  for (int v = 0; v < vPerB; ++v) acc += (float)p[(long)v * C];
  s[b * C + c] = acc;
}

__global__ void k_se_mlp(const float* __restrict__ s, const float* __restrict__ mask, int ncell,
                         const float* __restrict__ w1, const float* __restrict__ b1,
                         const float* __restrict__ w2, const float* __restrict__ b2,
                         float* __restrict__ g, int C, int R) {
  int b = blockIdx.x, t = threadIdx.x;
  __shared__ float cnt;
  __shared__ float sv[512];
  __shared__ float h[64];
  if (t == 0) {
    float c = 0.f;
    const float* mp = mask + (long)b * ncell;
    for (int i = 0; i < ncell; ++i) c += mp[i];
    cnt = fmaxf(c, 1.0f);
  }
  __syncthreads();
  if (t < C) sv[t] = s[b * C + t] / cnt;
  __syncthreads();
  if (t < R) {
    float a = b1[t];
    for (int c = 0; c < C; ++c) a += sv[c] * w1[c * R + t];
    h[t] = fmaxf(a, 0.f);
  }
  __syncthreads();
  if (t < C) {
    float a = b2[t];
    for (int r = 0; r < R; ++r) a += h[r] * w2[r * C + t];
    g[b * C + t] = 1.f / (1.f + expf(-a));
  }
}

__global__ __launch_bounds__(256) void k_se_apply(const _Float16* __restrict__ o,
                                                  const float* __restrict__ g,
                                                  const _Float16* __restrict__ xr,
                                                  _Float16* __restrict__ out,
                                                  int vPerB, int C, long total) {
  long i = (long)blockIdx.x * blockDim.x + threadIdx.x;
  if (i >= total) return;
  int c = (int)(i % C);
  int b = (int)(i / ((long)vPerB * C));
  float v = (float)o[i] * g[b * C + c] + (float)xr[i];
  out[i] = (_Float16)fmaxf(v, 0.f);
}

__global__ __launch_bounds__(256) void k_gather(const _Float16* __restrict__ src,
                                                const int* __restrict__ vc,
                                                float* __restrict__ out, int N, int C) {
  int i = blockIdx.x * blockDim.x + threadIdx.x;
  if (i >= N * C) return;
  int vi = i / C, c = i % C;
  int b = vc[vi * 4 + 0], z = vc[vi * 4 + 1], y = vc[vi * 4 + 2], x = vc[vi * 4 + 3];
  long vox = ((long)(b * 8 + z) * 48 + y) * 48 + x;
  out[i] = (float)src[vox * C + c];
}

// ---------------------------------------------------------------------------
// Implicit-GEMM 3x3x3 conv. One wave32 computes a 16(vox) x 64(cout) tile:
// 4 f32 accumulators, A fragment reused across 4 WMMAs per 32-wide K chunk.
// CIN is a template parameter so the chunk loop fully unrolls and all K
// addressing becomes immediate offsets.
// MODE 0: stride1 pad1; MODE 1: stride2 pad1 (down); MODE 2: transposed x2 (up).
// ep 0: bias+BN*mask; ep 1: + ReLU; ep 2: + residual, ReLU.
// ---------------------------------------------------------------------------

struct ConvArgs {
  const _Float16* X;     // [B,Di,Hi,Wi,Cin] f16, Cin % 32 == 0
  const _Float16* Wt;    // [Cout][27*Cin] f16 (pre-flipped for MODE 2)
  _Float16* Y;           // [B,Do,Ho,Wo,Cout] f16
  const float* bias;     // [Cout] or null
  const float *g, *bb, *mm, *vv;  // BN gamma/beta/mean/var [Cout]
  const float* mask;     // [B,Do,Ho,Wo]
  const _Float16* resid; // [B,Do,Ho,Wo,Cout] or null (ep==2)
  int Bn, Di, Hi, Wi, Cin, Do, Ho, Wo, Cout, ep;
};

template <int MODE, int CIN>
__global__ __launch_bounds__(32) void k_conv_wmma(ConvArgs a) {
  const int lane = threadIdx.x;
  const int row  = lane & 15;
  const bool hi  = lane >= 16;
  const int Mtot = a.Bn * a.Do * a.Ho * a.Wo;
  const int m = blockIdx.x * 16 + row;
  const bool mval = m < Mtot;
  int b, z, y, x;
  {
    int t = mval ? m : 0;
    x = t % a.Wo; t /= a.Wo;
    y = t % a.Ho; t /= a.Ho;
    z = t % a.Do; b = t / a.Do;
  }
  constexpr long Ktot = 27L * CIN;
  // B fragment: lane holds column n, K rows {0..7,16..23} / {8..15,24..31} of each 32-chunk,
  // in the [Cout][Ktot] packed layout. 4 N-tiles per wave.
  const int nbase = blockIdx.y * 64 + row;
  const _Float16* wp = a.Wt + (long)nbase * Ktot + (hi ? 8 : 0);
  constexpr long WSTR = 16L * Ktot;  // row stride between N-tiles
  // A fragment: lanes L / L+16 both hold voxel row L with K {0..7,16..23} / {8..15,24..31}.
  const int klo = hi ? 8 : 0;
  v8f acc[4] = {{}, {}, {}, {}};
  for (int t = 0; t < 27; ++t) {
    const int tz = t / 9, ty = (t / 3) % 3, tx = t % 3;
    int iz, iy, ix;
    bool valid = mval;
    if (MODE == 0)      { iz = z + tz - 1;     iy = y + ty - 1;     ix = x + tx - 1; }
    else if (MODE == 1) { iz = 2 * z + tz - 1; iy = 2 * y + ty - 1; ix = 2 * x + tx - 1; }
    else {
      int dz = z + tz - 1, dy = y + ty - 1, dx = x + tx - 1;
      valid = valid && !((dz | dy | dx) & 1);
      iz = dz >> 1; iy = dy >> 1; ix = dx >> 1;
    }
    valid = valid && (unsigned)iz < (unsigned)a.Di && (unsigned)iy < (unsigned)a.Hi &&
            (unsigned)ix < (unsigned)a.Wi;
    const _Float16* ap = a.X + (long)(((b * a.Di + iz) * a.Hi + iy) * a.Wi + ix) * CIN;
    const long kb = (long)t * CIN;
#pragma unroll
    for (int cc = 0; cc < CIN / 32; ++cc) {
      v8h a0 = {}, a1 = {};
      if (valid) {
        const _Float16* p0 = ap + cc * 32 + klo;
        a0 = *(const v8h*)p0;          // 16B contiguous
        a1 = *(const v8h*)(p0 + 16);   // 16B contiguous
      }
      v16h A = __builtin_shufflevector(a0, a1, 0, 1, 2, 3, 4, 5, 6, 7,
                                       8, 9, 10, 11, 12, 13, 14, 15);
      const _Float16* q = wp + kb + cc * 32;
#pragma unroll
      for (int nt = 0; nt < 4; ++nt) {
        const _Float16* qn = q + nt * WSTR;
        v8h b0 = *(const v8h*)qn;
        v8h b1 = *(const v8h*)(qn + 16);
        v16h Bm = __builtin_shufflevector(b0, b1, 0, 1, 2, 3, 4, 5, 6, 7,
                                          8, 9, 10, 11, 12, 13, 14, 15);
        acc[nt] = __builtin_amdgcn_wmma_f32_16x16x32_f16(false, A, false, Bm,
                                                         (short)0, acc[nt], false, false);
      }
    }
  }
  // Fused epilogue. C/D layout: lane -> channel n, VGPR r -> voxel (hi?8:0)+r.
  const int m0 = blockIdx.x * 16 + (hi ? 8 : 0);
#pragma unroll
  for (int nt = 0; nt < 4; ++nt) {
    const int n = nbase + nt * 16;
    float bs   = a.bias ? a.bias[n] : 0.0f;
    float sc   = a.g[n] * rsqrtf(a.vv[n] + 1e-3f);
    float mean = a.mm[n];
    float beta = a.bb[n];
#pragma unroll
    for (int r = 0; r < 8; ++r) {
      int mo = m0 + r;
      if (mo >= Mtot) break;
      float v = (acc[nt][r] + bs - mean) * sc + beta;
      v *= a.mask[mo];
      long o = (long)mo * a.Cout + n;
      if (a.ep == 2) v += (float)a.resid[o];
      if (a.ep != 0) v = fmaxf(v, 0.0f);
      a.Y[o] = (_Float16)v;
    }
  }
}

template <int MODE>
static void launch_conv_t(const ConvArgs& a, dim3 g, hipStream_t s) {
  switch (a.Cin) {
    case 32:   k_conv_wmma<MODE, 32><<<g, 32, 0, s>>>(a);   break;
    case 64:   k_conv_wmma<MODE, 64><<<g, 32, 0, s>>>(a);   break;
    case 128:  k_conv_wmma<MODE, 128><<<g, 32, 0, s>>>(a);  break;
    case 256:  k_conv_wmma<MODE, 256><<<g, 32, 0, s>>>(a);  break;
    case 512:  k_conv_wmma<MODE, 512><<<g, 32, 0, s>>>(a);  break;
    case 1024: k_conv_wmma<MODE, 1024><<<g, 32, 0, s>>>(a); break;
    default: break;
  }
}

// ---------------------------------------------------------------------------
// Host orchestration
// ---------------------------------------------------------------------------

extern "C" void kernel_launch(void* const* d_in, const int* in_sizes, int n_in,
                              void* d_out, int out_size, void* d_ws, size_t ws_size,
                              hipStream_t stream) {
  (void)n_in; (void)out_size; (void)ws_size;
  const float* vfeat = (const float*)d_in[0];
  const int*   vcoor = (const int*)d_in[1];
  const int Nvox = in_sizes[1] / 4;

  // ---- parameter leaves: pytree (alphabetical dict) order after 3 inputs ----
  int cur = 3;
  auto nxt = [&]() { return (const float*)d_in[cur++]; };
  struct BNP { const float *b, *g, *m, *v; };
  struct CMP { const float* b; BNP bn; const float* w; };
  struct BBP { BNP bn1, bn2; const float *c1b, *c1w, *c2b, *c2w, *se_b1, *se_b2, *se_w1, *se_w2; };
  auto rdBN = [&]() { BNP p; p.b = nxt(); p.g = nxt(); p.m = nxt(); p.v = nxt(); return p; };
  auto rdCM = [&]() { CMP p; p.b = nxt(); p.bn = rdBN(); p.w = nxt(); return p; };
  auto rdBB = [&](bool se) {
    BBP p{}; p.bn1 = rdBN(); p.bn2 = rdBN();
    p.c1b = nxt(); p.c1w = nxt(); p.c2b = nxt(); p.c2w = nxt();
    if (se) { p.se_b1 = nxt(); p.se_b2 = nxt(); p.se_w1 = nxt(); p.se_w2 = nxt(); }
    return p;
  };
  CMP pctx = rdCM();
  BBP c1[2] = { rdBB(false), rdBB(false) };
  BBP c2[2] = { rdBB(false), rdBB(false) };
  BBP c3[2] = { rdBB(false), rdBB(true) };
  BBP c4[2] = { rdBB(false), rdBB(true) };
  BNP ci_bn = rdBN(); const float* ci_w = nxt();
  CMP dn2 = rdCM(), dn3 = rdCM(), dn4 = rdCM();
  struct UPP { CMP btl, out; BBP t; };
  auto rdUP = [&]() { UPP u; u.btl = rdCM(); u.out = rdCM(); u.t = rdBB(false); return u; };
  UPP up1 = rdUP(), up2 = rdUP(), up3 = rdUP(), up4 = rdUP();

  // ---- workspace bump allocator (deterministic) ----
  size_t off = 0;
  auto alloc  = [&](size_t bytes) { void* p = (char*)d_ws + off; off += (bytes + 255) & ~(size_t)255; return p; };
  auto allocH = [&](size_t elems) { return (_Float16*)alloc(elems * sizeof(_Float16)); };
  auto allocF = [&](size_t elems) { return (float*)alloc(elems * sizeof(float)); };

  const int B = 2;
  const int D1 = 8, H1 = 48, W1 = 48; const int V1 = B * D1 * H1 * W1;
  const int D2 = 4, H2 = 24, W2 = 24; const int V2 = B * D2 * H2 * W2;
  const int D3 = 2, H3 = 12, W3 = 12; const int V3 = B * D3 * H3 * W3;
  const int D4 = 1, H4 = 6,  W4 = 6;  const int V4 = B * D4 * H4 * W4;

  _Float16* x0 = allocH((size_t)V1 * 32);
  float* m1 = allocF(V1); float* m2 = allocF(V2);
  float* m3 = allocF(V3); float* m4 = allocF(V4);
  _Float16* x1  = allocH((size_t)V1 * 64);
  _Float16* t1  = allocH((size_t)V1 * 64);
  _Float16* x2  = allocH((size_t)V2 * 128);
  _Float16* t2  = allocH((size_t)V2 * 128);
  _Float16* x3  = allocH((size_t)V3 * 256);
  _Float16* t3  = allocH((size_t)V3 * 256);
  _Float16* o3  = allocH((size_t)V3 * 256);
  _Float16* x4  = allocH((size_t)V4 * 512);
  _Float16* t4  = allocH((size_t)V4 * 512);
  _Float16* o4  = allocH((size_t)V4 * 512);
  _Float16* tb4 = allocH((size_t)V4 * 512);
  _Float16* cat4 = allocH((size_t)V4 * 1024);
  _Float16* btl4 = allocH((size_t)V4 * 512);
  _Float16* u4   = allocH((size_t)V3 * 256);
  _Float16* cat3 = allocH((size_t)V3 * 512);
  _Float16* btl3 = allocH((size_t)V3 * 256);
  _Float16* u3   = allocH((size_t)V2 * 128);
  _Float16* cat2 = allocH((size_t)V2 * 256);
  _Float16* btl2 = allocH((size_t)V2 * 128);
  _Float16* u2   = allocH((size_t)V1 * 64);
  _Float16* cat1 = allocH((size_t)V1 * 128);
  _Float16* btl1 = allocH((size_t)V1 * 64);
  _Float16* u1   = allocH((size_t)V1 * 64);
  _Float16* xc   = allocH((size_t)V1 * 64);
  float* sbuf = allocF((size_t)B * 512);
  float* gbuf = allocF((size_t)B * 512);

  // ---- densify + masks ----
  {
    long n32 = (long)V1 * 32 / 2;
    k_fill_zero<<<(int)((n32 + 255) / 256), 256, 0, stream>>>((uint32_t*)x0, n32);
    long nm = V1;
    k_fill_zero<<<(int)((nm + 255) / 256), 256, 0, stream>>>((uint32_t*)m1, nm);
  }
  k_scatter<<<(Nvox + 255) / 256, 256, 0, stream>>>(vfeat, vcoor, x0, m1, Nvox);
  k_maxpool_mask<<<(V2 + 255) / 256, 256, 0, stream>>>(m1, m2, B, D1, H1, W1, D2, H2, W2);
  k_maxpool_mask<<<(V3 + 255) / 256, 256, 0, stream>>>(m2, m3, B, D2, H2, W2, D3, H3, W3);
  k_maxpool_mask<<<(V4 + 255) / 256, 256, 0, stream>>>(m3, m4, B, D3, H3, W3, D4, H4, W4);

  // ---- helpers ----
  auto pack = [&](const float* w, int cin, int cout, int cinP, bool flip) {
    _Float16* wt = allocH((size_t)27 * cinP * cout);
    long total = (long)27 * cinP * cout;
    k_pack_w<<<(int)((total + 255) / 256), 256, 0, stream>>>(w, wt, cin, cout, cinP, flip ? 1 : 0);
    return wt;
  };
  auto conv = [&](const _Float16* X, const _Float16* Wt, _Float16* Y,
                  const float* bias, const BNP& bn, const float* mask, const _Float16* resid,
                  int Di, int Hi, int Wi, int Cin, int Do, int Ho, int Wo, int Cout,
                  int mode, int ep) {
    ConvArgs a;
    a.X = X; a.Wt = Wt; a.Y = Y; a.bias = bias;
    a.g = bn.g; a.bb = bn.b; a.mm = bn.m; a.vv = bn.v;
    a.mask = mask; a.resid = resid;
    a.Bn = B; a.Di = Di; a.Hi = Hi; a.Wi = Wi; a.Cin = Cin;
    a.Do = Do; a.Ho = Ho; a.Wo = Wo; a.Cout = Cout; a.ep = ep;
    int Mtot = B * Do * Ho * Wo;
    dim3 g((Mtot + 15) / 16, Cout / 64);  // 16 x 64 tile per wave
    if (mode == 0)      launch_conv_t<0>(a, g, stream);
    else if (mode == 1) launch_conv_t<1>(a, g, stream);
    else                launch_conv_t<2>(a, g, stream);
  };
  auto block = [&](const _Float16* xin, _Float16* tmp, _Float16* out, const BBP& p,
                   const float* mask, int D, int H, int W, int C, _Float16* obuf) {
    _Float16* w1 = pack(p.c1w, C, C, C, false);
    conv(xin, w1, tmp, p.c1b, p.bn1, mask, nullptr, D, H, W, C, D, H, W, C, 0, 1);
    _Float16* w2 = pack(p.c2w, C, C, C, false);
    int vPerB = D * H * W;
    if (!p.se_w1) {
      conv(tmp, w2, out, p.c2b, p.bn2, mask, xin, D, H, W, C, D, H, W, C, 0, 2);
    } else {
      conv(tmp, w2, obuf, p.c2b, p.bn2, mask, nullptr, D, H, W, C, D, H, W, C, 0, 0);
      k_se_reduce<<<dim3((C + 63) / 64, B), 64, 0, stream>>>(obuf, sbuf, vPerB, C);
      k_se_mlp<<<B, C, 0, stream>>>(sbuf, mask, vPerB, p.se_w1, p.se_b1, p.se_w2, p.se_b2,
                                    gbuf, C, C / 16);
      long tot = (long)B * vPerB * C;
      k_se_apply<<<(int)((tot + 255) / 256), 256, 0, stream>>>(obuf, gbuf, xin, out, vPerB, C, tot);
    }
  };
  auto concat = [&](const _Float16* A, const _Float16* Bs, _Float16* out, int nvox, int C) {
    long tot = (long)nvox * 2 * C;
    k_concat<<<(int)((tot + 255) / 256), 256, 0, stream>>>(A, Bs, out, C, tot);
  };

  // ---- encoder ----
  _Float16* wci = pack(ci_w, 4, 64, 32, false);
  conv(x0, wci, x1, nullptr, ci_bn, m1, nullptr, D1, H1, W1, 32, D1, H1, W1, 64, 0, 1);
  block(x1, t1, x1, c1[0], m1, D1, H1, W1, 64, nullptr);
  block(x1, t1, x1, c1[1], m1, D1, H1, W1, 64, nullptr);

  _Float16* wd2 = pack(dn2.w, 64, 128, 64, false);
  conv(x1, wd2, x2, dn2.b, dn2.bn, m2, nullptr, D1, H1, W1, 64, D2, H2, W2, 128, 1, 1);
  block(x2, t2, x2, c2[0], m2, D2, H2, W2, 128, nullptr);
  block(x2, t2, x2, c2[1], m2, D2, H2, W2, 128, nullptr);

  _Float16* wd3 = pack(dn3.w, 128, 256, 128, false);
  conv(x2, wd3, x3, dn3.b, dn3.bn, m3, nullptr, D2, H2, W2, 128, D3, H3, W3, 256, 1, 1);
  block(x3, t3, x3, c3[0], m3, D3, H3, W3, 256, nullptr);
  block(x3, t3, x3, c3[1], m3, D3, H3, W3, 256, o3);

  _Float16* wd4 = pack(dn4.w, 256, 512, 256, false);
  conv(x3, wd4, x4, dn4.b, dn4.bn, m4, nullptr, D3, H3, W3, 256, D4, H4, W4, 512, 1, 1);
  block(x4, t4, x4, c4[0], m4, D4, H4, W4, 512, nullptr);
  block(x4, t4, x4, c4[1], m4, D4, H4, W4, 512, o4);

  // ---- decoder ----
  // up4 (skip = x4 itself, so block goes out-of-place)
  block(x4, t4, tb4, up4.t, m4, D4, H4, W4, 512, nullptr);
  concat(x4, tb4, cat4, V4, 512);
  _Float16* wb4 = pack(up4.btl.w, 1024, 512, 1024, false);
  conv(cat4, wb4, btl4, up4.btl.b, up4.btl.bn, m4, nullptr, D4, H4, W4, 1024, D4, H4, W4, 512, 0, 1);
  _Float16* wo4 = pack(up4.out.w, 512, 256, 512, true);
  conv(btl4, wo4, u4, up4.out.b, up4.out.bn, m3, nullptr, D4, H4, W4, 512, D3, H3, W3, 256, 2, 1);

  // up3
  block(u4, t3, u4, up3.t, m3, D3, H3, W3, 256, nullptr);
  concat(x3, u4, cat3, V3, 256);
  _Float16* wb3 = pack(up3.btl.w, 512, 256, 512, false);
  conv(cat3, wb3, btl3, up3.btl.b, up3.btl.bn, m3, nullptr, D3, H3, W3, 512, D3, H3, W3, 256, 0, 1);
  _Float16* wo3 = pack(up3.out.w, 256, 128, 256, true);
  conv(btl3, wo3, u3, up3.out.b, up3.out.bn, m2, nullptr, D3, H3, W3, 256, D2, H2, W2, 128, 2, 1);

  // up2
  block(u3, t2, u3, up2.t, m2, D2, H2, W2, 128, nullptr);
  concat(x2, u3, cat2, V2, 128);
  _Float16* wb2 = pack(up2.btl.w, 256, 128, 256, false);
  conv(cat2, wb2, btl2, up2.btl.b, up2.btl.bn, m2, nullptr, D2, H2, W2, 256, D2, H2, W2, 128, 0, 1);
  _Float16* wo2 = pack(up2.out.w, 128, 64, 128, true);
  conv(btl2, wo2, u2, up2.out.b, up2.out.bn, m1, nullptr, D2, H2, W2, 128, D1, H1, W1, 64, 2, 1);

  // up1 (no transposed conv)
  block(u2, t1, u2, up1.t, m1, D1, H1, W1, 64, nullptr);
  concat(x1, u2, cat1, V1, 64);
  _Float16* wb1 = pack(up1.btl.w, 128, 64, 128, false);
  conv(cat1, wb1, btl1, up1.btl.b, up1.btl.bn, m1, nullptr, D1, H1, W1, 128, D1, H1, W1, 64, 0, 1);
  _Float16* wo1 = pack(up1.out.w, 64, 64, 64, false);
  conv(btl1, wo1, u1, up1.out.b, up1.out.bn, m1, nullptr, D1, H1, W1, 64, D1, H1, W1, 64, 0, 1);

  // context + gather
  _Float16* wcx = pack(pctx.w, 64, 64, 64, false);
  conv(u1, wcx, xc, pctx.b, pctx.bn, m1, nullptr, D1, H1, W1, 64, D1, H1, W1, 64, 0, 1);
  k_gather<<<(Nvox * 64 + 255) / 256, 256, 0, stream>>>(xc, vcoor, (float*)d_out, Nvox, 64);
}
